// Attention_46145128628680
// MI455X (gfx1250) — matrix operand order repaired and checked by reading, not compiled, not run on hardware
//
#include <hip/hip_runtime.h>

#define SEQ 8192
#define DIM 1024

typedef __attribute__((ext_vector_type(8)))  __bf16 v8bf;
typedef __attribute__((ext_vector_type(16))) __bf16 v16bf;
typedef __attribute__((ext_vector_type(8)))  float  v8f;

union FragAB { v16bf f; v8bf h[2]; };

__device__ __forceinline__ unsigned short f2bf(float x) {
  unsigned int u = __float_as_uint(x);
  u += 0x7fffu + ((u >> 16) & 1u);            // round-to-nearest-even
  return (unsigned short)(u >> 16);
}

// A-matrix 16x32 bf16 fragment: lane l holds row l%16,
// K chunks [k0 + 8*lh, +8) and [k0 + 16 + 8*lh, +8)   (ISA 7.12.2)
__device__ __forceinline__ v16bf load_frag_a(const unsigned short* base, int lh) {
  FragAB fr;
  fr.h[0] = *(const v8bf*)(base + lh * 8);
  fr.h[1] = *(const v8bf*)(base + 16 + lh * 8);
  return fr.f;
}

// B-matrix 32x16 bf16 fragment: lane l holds column l%16,
// K chunks [k0 + 8*lh, +8) and [k0 + 16 + 8*lh, +8)   (ISA 7.12.4 B layout)
__device__ __forceinline__ v16bf load_frag_b(const unsigned short* base, int lh) {
  FragAB fr;
  fr.h[0] = *(const v8bf*)(base + lh * 8);
  fr.h[1] = *(const v8bf*)(base + 16 + lh * 8);
  return fr.f;
}

__device__ __forceinline__ v8f wmma_bf16(v16bf a, v16bf b, v8f c) {
  return __builtin_amdgcn_wmma_f32_16x16x32_bf16(false, a, false, b, (short)0, c,
                                                 false, false);
}

// ---------------------------------------------------------------- fp32 -> bf16
__global__ void cvt_f32_bf16(const float* __restrict__ src,
                             unsigned short* __restrict__ dst, int n) {
  int i = blockIdx.x * blockDim.x + threadIdx.x;
  int stride = gridDim.x * blockDim.x;
  for (; i < n; i += stride) dst[i] = f2bf(src[i]);
}

// ---------------------------------------------------------------- y = x @ W^T
// grid: (SEQ/16, 2); block 256 (8 waves). Wave w covers cols [by*512+w*64, +64).
// transpose==0: out[m][n] bf16 (scaled). transpose==1: out[n][m] (V^T).
__global__ void __launch_bounds__(256)
qkv_gemm(const unsigned short* __restrict__ xb,
         const unsigned short* __restrict__ wb,
         unsigned short* __restrict__ out, float scale, int transpose) {
  const int tid = threadIdx.x;
  const int wave = tid >> 5, lane = tid & 31;
  const int lm = lane & 15, lh = lane >> 4;
  const int m0 = blockIdx.x * 16;
  const int n0 = blockIdx.y * 512 + wave * 64;

  v8f c[4];
#pragma unroll
  for (int t = 0; t < 4; ++t)
#pragma unroll
    for (int e = 0; e < 8; ++e) c[t][e] = 0.f;

  const unsigned short* arow = xb + (size_t)(m0 + lm) * DIM;
  for (int d0 = 0; d0 < DIM; d0 += 32) {
    v16bf a = load_frag_a(arow + d0, lh);
#pragma unroll
    for (int t = 0; t < 4; ++t) {
      const unsigned short* brow = wb + (size_t)(n0 + t * 16 + lm) * DIM + d0;
      v16bf b = load_frag_b(brow, lh);
      c[t] = wmma_bf16(a, b, c[t]);
    }
  }

  // C layout: lane l, vgpr r -> row m0 + r + 8*lh, col n0 + t*16 + l%16
  if (!transpose) {
#pragma unroll
    for (int t = 0; t < 4; ++t)
#pragma unroll
      for (int r = 0; r < 8; ++r)
        out[(size_t)(m0 + r + 8 * lh) * DIM + (n0 + t * 16 + lm)] =
            f2bf(c[t][r] * scale);
  } else {
#pragma unroll
    for (int t = 0; t < 4; ++t)
#pragma unroll
      for (int r = 0; r < 8; ++r)
        out[(size_t)(n0 + t * 16 + lm) * SEQ + (m0 + r + 8 * lh)] =
            f2bf(c[t][r] * scale);
  }
}

// ---------------------------------------------------------------- flash attn
// grid: SEQ/16 blocks of 256 threads. Block owns 16 query rows; wave w owns
// output columns [w*128, +128). 256 keys per iteration (32 per wave).
__global__ void __launch_bounds__(256)
attn_kernel(const unsigned short* __restrict__ qb,
            const unsigned short* __restrict__ kb,
            const unsigned short* __restrict__ vt,
            float* __restrict__ out) {
  __shared__ __align__(16) unsigned short Qs[16 * DIM];   // 32 KB
  __shared__ float Ss[8][16][32];                         // 16 KB scores
  __shared__ __align__(16) unsigned short Ps[16][256];    // 8 KB probs (bf16)
  __shared__ float rowmax_part[8][16];
  __shared__ float rowsum_part[8][16];
  __shared__ float m_state[16], l_state[16];
  __shared__ float mnew_arr[16], alpha_arr[16];

  const int tid = threadIdx.x;
  const int wave = tid >> 5, lane = tid & 31;
  const int lm = lane & 15, lh = lane >> 4;
  const int q0 = blockIdx.x * 16;

  {  // stage Q tile (already scaled by 1/sqrt(d)) into LDS
    const uint4* src = (const uint4*)(qb + (size_t)q0 * DIM);
    uint4* dst = (uint4*)Qs;
    for (int i = tid; i < (16 * DIM) / 8; i += 256) dst[i] = src[i];
  }
  if (tid < 16) { m_state[tid] = -1e30f; l_state[tid] = 0.f; }
  __syncthreads();

  v8f O[8];
#pragma unroll
  for (int t = 0; t < 8; ++t)
#pragma unroll
    for (int e = 0; e < 8; ++e) O[t][e] = 0.f;

  const int kend = q0 + 15;
  for (int kbase = 0; kbase <= kend; kbase += 256) {
    const int kw = kbase + wave * 32;  // this wave's 32 keys

    // ---- Phase 1: S(16x32) = Q @ K^T over full depth 1024 (64 WMMAs) ----
    v8f s0, s1;
#pragma unroll
    for (int e = 0; e < 8; ++e) { s0[e] = 0.f; s1[e] = 0.f; }
    const unsigned short* qrow  = Qs + lm * DIM;
    const unsigned short* krow0 = kb + (size_t)(kw + lm) * DIM;
    const unsigned short* krow1 = kb + (size_t)(kw + 16 + lm) * DIM;
    for (int d0 = 0; d0 < DIM; d0 += 32) {
      v16bf a  = load_frag_a(qrow + d0, lh);
      v16bf b0 = load_frag_b(krow0 + d0, lh);
      v16bf b1 = load_frag_b(krow1 + d0, lh);
      s0 = wmma_bf16(a, b0, s0);
      s1 = wmma_bf16(a, b1, s1);
    }
#pragma unroll
    for (int r = 0; r < 8; ++r) {
      Ss[wave][r + 8 * lh][lm]      = s0[r];
      Ss[wave][r + 8 * lh][16 + lm] = s1[r];
    }
    // per-row chunk max (causal-masked), same-wave LDS read-after-write
    if (lh == 0) {
      const int qr = q0 + lm;
      float cmax = -1e30f;
#pragma unroll
      for (int j = 0; j < 32; ++j) {
        float s = Ss[wave][lm][j];
        cmax = fmaxf(cmax, (kw + j <= qr) ? s : -1e30f);
      }
      rowmax_part[wave][lm] = cmax;
    }
    __syncthreads();  // B1

    // ---- Phase 2: global row max, exp, P(bf16) to LDS, partial sums ----
    if (lh == 0) {
      float mnew = m_state[lm];
#pragma unroll
      for (int w = 0; w < 8; ++w) mnew = fmaxf(mnew, rowmax_part[w][lm]);
      const int qr = q0 + lm;
      float rsum = 0.f;
#pragma unroll
      for (int j = 0; j < 32; ++j) {
        float s = Ss[wave][lm][j];
        float p = (kw + j <= qr) ? __expf(s - mnew) : 0.f;
        rsum += p;
        Ps[lm][wave * 32 + j] = f2bf(p);
      }
      rowsum_part[wave][lm] = rsum;
      if (wave == 0) mnew_arr[lm] = mnew;
    }
    __syncthreads();  // B2

    // ---- Phase 3: softmax running-state update ----
    if (tid < 16) {
      float mold = m_state[tid];
      float mnew = mnew_arr[tid];
      float alpha = __expf(mold - mnew);
      float lsum = alpha * l_state[tid];
#pragma unroll
      for (int w = 0; w < 8; ++w) lsum += rowsum_part[w][tid];
      alpha_arr[tid] = alpha;
      m_state[tid] = mnew;
      l_state[tid] = lsum;
    }
    __syncthreads();  // B3

    // ---- Phase 4: rescale O, then O += P(16x256) @ V(256x128) (64 WMMAs) ----
    float al[8];
#pragma unroll
    for (int r = 0; r < 8; ++r) al[r] = alpha_arr[r + 8 * lh];
#pragma unroll
    for (int t = 0; t < 8; ++t)
#pragma unroll
      for (int r = 0; r < 8; ++r) O[t][r] *= al[r];

    const unsigned short* prow = Ps[lm];
#pragma unroll
    for (int ks = 0; ks < 8; ++ks) {
      v16bf a = load_frag_a(prow + ks * 32, lh);
#pragma unroll
      for (int t = 0; t < 8; ++t) {
        const unsigned short* vrow =
            vt + (size_t)(wave * 128 + t * 16 + lm) * SEQ + kbase + ks * 32;
        v16bf b = load_frag_b(vrow, lh);
        O[t] = wmma_bf16(a, b, O[t]);
      }
    }
    __syncthreads();  // B4 (protect Ps/Ss/partials for next iteration)
  }

  // ---- Epilogue: O /= l, write fp32 ----
  float li[8];
#pragma unroll
  for (int r = 0; r < 8; ++r) li[r] = 1.0f / l_state[r + 8 * lh];
#pragma unroll
  for (int t = 0; t < 8; ++t)
#pragma unroll
    for (int r = 0; r < 8; ++r)
      out[(size_t)(q0 + r + 8 * lh) * DIM + wave * 128 + t * 16 + lm] =
          O[t][r] * li[r];
}

// ---------------------------------------------------------------- launcher
extern "C" void kernel_launch(void* const* d_in, const int* in_sizes, int n_in,
                              void* d_out, int out_size, void* d_ws,
                              size_t ws_size, hipStream_t stream) {
  const float* x  = (const float*)d_in[0];
  const float* wq = (const float*)d_in[1];
  const float* wk = (const float*)d_in[2];
  const float* wv = (const float*)d_in[3];
  float* out = (float*)d_out;

  unsigned short* ws  = (unsigned short*)d_ws;
  unsigned short* xb  = ws;                               // [SEQ x DIM] bf16
  unsigned short* qb  = xb  + (size_t)SEQ * DIM;          // Q (pre-scaled)
  unsigned short* kbp = qb  + (size_t)SEQ * DIM;          // K
  unsigned short* vtb = kbp + (size_t)SEQ * DIM;          // V^T [DIM x SEQ]
  unsigned short* wqb = vtb + (size_t)SEQ * DIM;
  unsigned short* wkb = wqb + (size_t)DIM * DIM;
  unsigned short* wvb = wkb + (size_t)DIM * DIM;

  cvt_f32_bf16<<<1024, 256, 0, stream>>>(x,  xb,  SEQ * DIM);
  cvt_f32_bf16<<<256,  256, 0, stream>>>(wq, wqb, DIM * DIM);
  cvt_f32_bf16<<<256,  256, 0, stream>>>(wk, wkb, DIM * DIM);
  cvt_f32_bf16<<<256,  256, 0, stream>>>(wv, wvb, DIM * DIM);

  dim3 g(SEQ / 16, 2);
  qkv_gemm<<<g, 256, 0, stream>>>(xb, wqb, qb,  0.03125f, 0);  // Q * 1/sqrt(d)
  qkv_gemm<<<g, 256, 0, stream>>>(xb, wkb, kbp, 1.0f,     0);
  qkv_gemm<<<g, 256, 0, stream>>>(xb, wvb, vtb, 1.0f,     1);  // store V^T

  attn_kernel<<<SEQ / 16, 256, 0, stream>>>(qb, kbp, vtb, out);
}
